// RnnGenerator_74028056314208
// MI455X (gfx1250) — matrix operand order repaired and checked by reading, not compiled, not run on hardware
//
#include <hip/hip_runtime.h>

// ---------------------------------------------------------------------------
// Types for gfx1250 WMMA
// ---------------------------------------------------------------------------
typedef __bf16 bf16;
typedef __attribute__((ext_vector_type(16))) __bf16 v16bf;
typedef __attribute__((ext_vector_type(8)))  __bf16 v8bf;
typedef __attribute__((ext_vector_type(8)))  float  v8f;

union ABfrag { v16bf v; v8bf h[2]; };

#define C_DIM 1024
#define B_DIM 256
#define L_DIM 4
#define T_DIM 256

// ---------------------------------------------------------------------------
// Fragment loaders (wave32); A and B both use contiguous k order k = 16*(lane/16) + s
//
// A (16x32 bf16, M x K): lane L (0..15) holds row M=L, K = k0+0..15;
// lane L+16 holds row M=L, K = k0+16..31.
// ---------------------------------------------------------------------------
__device__ __forceinline__ v16bf load_a_frag(const bf16* __restrict__ act,
                                             int m0, int k0, int lane) {
    const int row = m0 + (lane & 15);
    const int hi  = lane >> 4;                 // 0 or 1
    const bf16* p = act + (size_t)row * C_DIM + k0 + 16 * hi;
    ABfrag u;
    u.h[0] = *(const v8bf*)(p);                // K chunk [k0+16hi .. +7]
    u.h[1] = *(const v8bf*)(p + 8);            // K chunk [k0+16hi+8 .. +7]
    return u.v;
}

// B (32x16 bf16, K x N): lane L (0..15) = column N, holds K = k0..k0+15
// (2 per VGPR); lane L+16 holds K = k0+16..k0+31.  W is row-major [c][k],
// and B[k][n] = W[n0+n][k0+k], so each lane reads 16 contiguous bf16.
__device__ __forceinline__ v16bf load_b_frag(const bf16* __restrict__ W,
                                             int n0, int k0, int lane) {
    const int col = n0 + (lane & 15);
    const int hi  = lane >> 4;
    const bf16* p = W + (size_t)col * C_DIM + k0 + 16 * hi;
    return *(const v16bf*)(p);
}

__device__ __forceinline__ v8f wmma_bf16(v16bf a, v16bf b, v8f c) {
    return __builtin_amdgcn_wmma_f32_16x16x32_bf16(
        /*neg_a=*/false, a, /*neg_b=*/false, b,
        /*c_mod=*/(short)0, c, /*reuse_a=*/false, /*reuse_b=*/false);
}

// C/D (16x16 f32): VGPR r, lanes 0..15 -> (M=r, N=lane), lanes 16..31 ->
// (M=r+8, N=lane-16).
__device__ __forceinline__ void store_tile(v8f acc, int m0, int n0, float bias,
                                           float* __restrict__ hf,
                                           bf16*  __restrict__ xout, int lane) {
    const int n  = n0 + (lane & 15);
    const int mb = m0 + (lane >> 4) * 8;
#pragma unroll
    for (int r = 0; r < 8; ++r) {
        float v = acc[r] + bias;
        v = v > 0.0f ? v : 0.0f;               // ReLU
        const size_t idx = (size_t)(mb + r) * C_DIM + n;
        hf[idx]   = v;
        xout[idx] = (bf16)v;
    }
}

// ---------------------------------------------------------------------------
// Fused GEMM stage: h = relu(A1 @ Wih^T + A2 @ Whh^T + bias_sum)
// M=256, N=1024, K=1024+1024.
// Wave tile 32x64 (2x4 WMMA tiles -> 8 WMMA per K-step, 12 b128 loads).
// 8 waves arranged 4(M) x 2(N) -> WG tile 128x128.  Grid (2, 8) = 16 WGs.
// #pragma unroll 2 caps in-flight loads => no VGPR spills.
// ---------------------------------------------------------------------------
__global__ __launch_bounds__(256)
void rnn_gemm_relu(const bf16* __restrict__ A1,   // xl        [256][1024]
                   const bf16* __restrict__ A2,   // hid_prev  [256][1024]
                   const bf16* __restrict__ Wih,  // [1024][1024] row-major c,k
                   const bf16* __restrict__ Whh,
                   const float* __restrict__ bias,// b_ih + b_hh  [1024]
                   float* __restrict__ hf,        // unnormalized h (f32)
                   bf16*  __restrict__ xout)      // unnormalized h (bf16)
{
    const int lane = threadIdx.x & 31;
    const int wave = threadIdx.x >> 5;
    const int m0 = blockIdx.x * 128 + (wave >> 1) * 32;   // 4 waves in M
    const int n0 = blockIdx.y * 128 + (wave & 1)  * 64;   // 2 waves in N

    v8f acc[2][4] = {};   // [mi][nj]

#pragma unroll
    for (int half = 0; half < 2; ++half) {
        const bf16* __restrict__ A = half ? A2  : A1;
        const bf16* __restrict__ W = half ? Whh : Wih;
#pragma unroll 2
        for (int k0 = 0; k0 < C_DIM; k0 += 32) {
            v16bf a0 = load_a_frag(A, m0,      k0, lane);
            v16bf a1 = load_a_frag(A, m0 + 16, k0, lane);
            v16bf b0 = load_b_frag(W, n0,      k0, lane);
            v16bf b1 = load_b_frag(W, n0 + 16, k0, lane);
            v16bf b2 = load_b_frag(W, n0 + 32, k0, lane);
            v16bf b3 = load_b_frag(W, n0 + 48, k0, lane);
            acc[0][0] = wmma_bf16(a0, b0, acc[0][0]);
            acc[1][0] = wmma_bf16(a1, b0, acc[1][0]);
            acc[0][1] = wmma_bf16(a0, b1, acc[0][1]);
            acc[1][1] = wmma_bf16(a1, b1, acc[1][1]);
            acc[0][2] = wmma_bf16(a0, b2, acc[0][2]);
            acc[1][2] = wmma_bf16(a1, b2, acc[1][2]);
            acc[0][3] = wmma_bf16(a0, b3, acc[0][3]);
            acc[1][3] = wmma_bf16(a1, b3, acc[1][3]);
        }
    }

#pragma unroll
    for (int j = 0; j < 4; ++j) {
        const float bj = bias[n0 + 16 * j + (lane & 15)];
        store_tile(acc[0][j], m0,      n0 + 16 * j, bj, hf, xout, lane);
        store_tile(acc[1][j], m0 + 16, n0 + 16 * j, bj, hf, xout, lane);
    }
}

// ---------------------------------------------------------------------------
// Per-step batch-dim L2 normalization.
// grid = (4 layers, 16 channel groups of 64), block = 256 threads.
// Thread (c = tid%64, bq = tid/64) sums 64 batch rows; 4 partials per channel
// reduced through LDS; then each thread normalizes its 64 rows.
// All global accesses coalesced (64 contiguous channels per quarter-block).
// ---------------------------------------------------------------------------
__global__ __launch_bounds__(256)
void rnn_normalize(const float* __restrict__ hf,   // [4][256][1024]
                   bf16*  __restrict__ hid,        // [4][256][1024]
                   bf16*  __restrict__ inp,        // [256][1024]
                   float* __restrict__ out,        // [256][256][1024]
                   int t)
{
    __shared__ float red[256];
    const int l  = blockIdx.x;
    const int c  = blockIdx.y * 64 + (threadIdx.x & 63);
    const int bq = threadIdx.x >> 6;                 // 0..3
    const float* h = hf + (size_t)l * B_DIM * C_DIM;

    float ss = 0.0f;
#pragma unroll 4
    for (int b = bq * 64; b < bq * 64 + 64; ++b) {
        float v = h[(size_t)b * C_DIM + c];
        ss += v * v;
    }
    red[threadIdx.x] = ss;
    __syncthreads();

    const int cl = threadIdx.x & 63;
    const float total = red[cl] + red[cl + 64] + red[cl + 128] + red[cl + 192];
    const float scale = 1.0f / (sqrtf(total) + 1e-12f);

    bf16* hl = hid + (size_t)l * B_DIM * C_DIM;
#pragma unroll 4
    for (int b = bq * 64; b < bq * 64 + 64; ++b) {
        const size_t idx = (size_t)b * C_DIM + c;
        float v = h[idx] * scale;
        hl[idx] = (bf16)v;
        if (l == 3) {
            inp[idx] = (bf16)v;
            out[((size_t)b * T_DIM + t) * C_DIM + c] = v;
        }
    }
}

// ---------------------------------------------------------------------------
// One-time prep: fp32 weights -> bf16, bias sum.
// ---------------------------------------------------------------------------
__global__ __launch_bounds__(256)
void rnn_convert(const float* __restrict__ Wih, const float* __restrict__ Whh,
                 const float* __restrict__ bih, const float* __restrict__ bhh,
                 bf16* __restrict__ Wihb, bf16* __restrict__ Whhb,
                 float* __restrict__ bsum)
{
    const size_t i = (size_t)blockIdx.x * blockDim.x + threadIdx.x;
    const size_t NW = (size_t)L_DIM * C_DIM * C_DIM;
    if (i < NW) {
        Wihb[i] = (bf16)Wih[i];
        Whhb[i] = (bf16)Whh[i];
    }
    if (i < (size_t)L_DIM * C_DIM) bsum[i] = bih[i] + bhh[i];
}

__global__ __launch_bounds__(256)
void rnn_init(const float* __restrict__ x, bf16* __restrict__ inp,
              bf16* __restrict__ hid)
{
    const size_t i = (size_t)blockIdx.x * blockDim.x + threadIdx.x; // 256*1024
    const size_t S = (size_t)B_DIM * C_DIM;
    inp[i]         = (bf16)0.0f;
    hid[i]         = (bf16)x[i];   // hid[0] = x
    hid[S + i]     = (bf16)0.0f;
    hid[2 * S + i] = (bf16)0.0f;
    hid[3 * S + i] = (bf16)0.0f;
}

// ---------------------------------------------------------------------------
// Launcher
// ---------------------------------------------------------------------------
extern "C" void kernel_launch(void* const* d_in, const int* in_sizes, int n_in,
                              void* d_out, int out_size, void* d_ws, size_t ws_size,
                              hipStream_t stream) {
    const float* x    = (const float*)d_in[0];
    const float* W_ih = (const float*)d_in[1];
    const float* W_hh = (const float*)d_in[2];
    const float* b_ih = (const float*)d_in[3];
    const float* b_hh = (const float*)d_in[4];
    float* out = (float*)d_out;

    const size_t WSZ = (size_t)C_DIM * C_DIM;     // per-layer weight elems
    const size_t HSZ = (size_t)B_DIM * C_DIM;     // per-layer state elems

    // Workspace carve-up (~24 MB total)
    char* ws = (char*)d_ws;
    bf16*  Wihb = (bf16*)(ws);                               // 8 MB
    bf16*  Whhb = (bf16*)(ws + (8ull << 20));                // 8 MB
    float* bsum = (float*)(ws + (16ull << 20));              // 16 KB
    bf16*  hid  = (bf16*)(ws + (16ull << 20) + (64ull << 10)); // 2 MB
    bf16*  inp  = hid + (size_t)L_DIM * HSZ;                 // 512 KB
    bf16*  xl0  = inp + HSZ;                                 // 512 KB
    bf16*  xl1  = xl0 + HSZ;                                 // 512 KB
    float* hf   = (float*)(xl1 + HSZ);                       // 4 MB

    // Prep (deterministic, re-run every call)
    {
        const size_t NW = (size_t)L_DIM * C_DIM * C_DIM;     // 4M
        dim3 grid((unsigned)((NW + 255) / 256));
        rnn_convert<<<grid, 256, 0, stream>>>(W_ih, W_hh, b_ih, b_hh,
                                              Wihb, Whhb, bsum);
        rnn_init<<<dim3((unsigned)(HSZ / 256)), 256, 0, stream>>>(x, inp, hid);
    }

    // A-input / x-output ping-pong across the 4 layers of a step
    const bf16* xin[L_DIM] = { inp, xl0, xl1, xl0 };
    bf16*       xo [L_DIM] = { xl0, xl1, xl0, xl1 };

    for (int t = 0; t < T_DIM; ++t) {
        for (int l = 0; l < L_DIM; ++l) {
            rnn_gemm_relu<<<dim3(2, 8), 256, 0, stream>>>(
                xin[l], hid + (size_t)l * HSZ,
                Wihb + (size_t)l * WSZ, Whhb + (size_t)l * WSZ,
                bsum + (size_t)l * C_DIM,
                hf + (size_t)l * HSZ, xo[l]);
        }
        rnn_normalize<<<dim3(4, 16), 256, 0, stream>>>(hf, hid, inp, out, t);
    }
}
